// EnhancedCGConv_47974784696411
// MI455X (gfx1250) — matrix operand order repaired and checked by reading, not compiled, not run on hardware
//
#include <hip/hip_runtime.h>
#include <hip/hip_bf16.h>

// EnhancedCGConv fused pipeline for MI455X (gfx1250, wave32, WMMA).
//
// Roofline: ~47 GFLOP vs ~0.4 GB of HBM traffic when fused (edge stream +
// L2-resident node table / segment accumulator) => must use the f16 WMMA
// path (16x16x32, f32 accumulate); the f32 WMMA shape (16x16x4) is 8x more
// instructions and would be compute-bound.

typedef __attribute__((ext_vector_type(16))) _Float16 v16h;
typedef __attribute__((ext_vector_type(8)))  float    v8f;

#define NNODES 50000
#define NEDGES 800000
#define FEAT   64
#define UNITS  64
#define NK     3
#define KU     192   // NK * UNITS

// ---------------------------------------------------------------- helpers

__device__ inline v8f wmma16(v16h a, v16h b, v8f c) {
  // D = A(16x32 f16) x B(32x16 f16) + C(16x16 f32)
  return __builtin_amdgcn_wmma_f32_16x16x32_f16(
      /*neg_a=*/false, a, /*neg_b=*/false, b,
      /*c_mod=*/(short)0, c, /*reuse_a=*/false, /*reuse_b=*/false);
}

// A-matrix (16x32, f16) lane fragment from row-major f32 memory.
// Lane l (0-15): row M=l, K = {0..7} u {16..23}; lanes 16-31: row M=l-16,
// K = {8..15} u {24..31}.  p must point at row_base + kbase + koff where
// koff = (lane>=16 ? 8 : 0).  Two contiguous 8-float runs -> b128 loads.
__device__ inline v16h load_a_f32(const float* __restrict__ p) {
  v16h a;
#pragma unroll
  for (int i = 0; i < 8; i++) { a[i] = (_Float16)p[i]; a[i + 8] = (_Float16)p[16 + i]; }
  return a;
}
__device__ inline v16h load_a_f32_scaled(const float* __restrict__ p, float s) {
  v16h a;
#pragma unroll
  for (int i = 0; i < 8; i++) { a[i] = (_Float16)(p[i] * s); a[i + 8] = (_Float16)(p[16 + i] * s); }
  return a;
}
__device__ inline v16h load_a_lds(const _Float16* p) {
  v16h a;
#pragma unroll
  for (int i = 0; i < 8; i++) { a[i] = p[i]; a[i + 8] = p[16 + i]; }
  return a;
}

// Pack a row-major f32 weight [Ktot, Ntot] into LDS as f16 B-matrix tiles.
// Tile t = kc*nChunks + nc covers K rows [kc*32, kc*32+32), N cols
// [nc*16, nc*16+16).  B lane layout (32x16 f16): lane l holds column
// n = l%16, K = kc*32 + (l/16)*8 + {0..7} u {16..23} -> 16 f16 per lane.
__device__ inline void pack_b(_Float16* dst, const float* __restrict__ W,
                              int Ntot, int nChunks, int tiles,
                              int tid, int nthr) {
  for (int p2 = tid; p2 < tiles * 32; p2 += nthr) {
    int t = p2 >> 5, lane = p2 & 31;
    int n  = ((t % nChunks) << 4) + (lane & 15);
    int k0 = ((t / nChunks) << 5) + ((lane >> 4) << 3);
    _Float16* d = dst + t * 512 + lane * 16;
#pragma unroll
    for (int i = 0; i < 16; i++) d[i] = (_Float16)W[(size_t)(k0 + i + (i >= 8 ? 8 : 0)) * Ntot + n];
  }
}
__device__ inline v16h bload(const _Float16* base, int t, int lane) {
  return *(const v16h*)(base + t * 512 + lane * 16);
}

// ---------------------------------------------------------------- kernels

__global__ void zero_ws_kernel(float* __restrict__ p, int n) {
  int i = blockIdx.x * blockDim.x + threadIdx.x;
  int stride = gridDim.x * blockDim.x;
  for (; i < n; i += stride) p[i] = 0.0f;
}

// One wave per 16-edge tile: edge MLP (2 GEMMs), gather src rows,
// K=3 message GEMMs, elementwise x edge weights, atomic scatter-add.
__global__ __launch_bounds__(128) void edge_msg_kernel(
    const float* __restrict__ node_feat, const int* __restrict__ edge_idx,
    const float* __restrict__ edge_feat, const float* __restrict__ Wk,
    const float* __restrict__ W1, const float* __restrict__ b1,
    const float* __restrict__ W2, const float* __restrict__ b2,
    float* __restrict__ sums, float* __restrict__ cnt) {
  __shared__ __align__(64) _Float16 sW1[8 * 512];    //  8 KB: W1  64x64
  __shared__ __align__(64) _Float16 sW2[24 * 512];   // 24 KB: W2  64x192
  __shared__ __align__(64) _Float16 sWk[24 * 512];   // 24 KB: Wk 3x64x64
  __shared__ __align__(64) _Float16 sH[4 * 512];     //  4 KB: 16x32 f16 per wave

  const int tid = threadIdx.x;
  pack_b(sW1, W1, UNITS, 4, 8, tid, blockDim.x);
  pack_b(sW2, W2, KU, 12, 24, tid, blockDim.x);
#pragma unroll
  for (int k = 0; k < NK; k++)
    pack_b(sWk + k * 8 * 512, Wk + k * FEAT * UNITS, UNITS, 4, 8, tid, blockDim.x);
  __syncthreads();

  const int wave = tid >> 5, lane = tid & 31;
  const int row  = lane & 15;           // A-matrix row / C-matrix column
  const int koff = (lane >> 4) << 3;    // A-layout K offset: 0 or 8
  const int moff = (lane >> 4) << 3;    // C-layout row offset: 0 or 8
  _Float16* myH = sH + wave * 512;

  const int nTiles = NEDGES / 16;
  const int waveG = blockIdx.x * (blockDim.x >> 5) + wave;
  const int waveStride = gridDim.x * (blockDim.x >> 5);

  for (int tile = waveG; tile < nTiles; tile += waveStride) {
    const int e0 = tile << 4;
    {  // prefetch next tile's edge features (gfx1250 global_prefetch)
      int pe = (tile + waveStride) << 4;
      if (pe < NEDGES) __builtin_prefetch(edge_feat + (size_t)pe * FEAT + lane * 32, 0, 0);
    }

    // A: edge feature tile (16x64)
    const float* efr = edge_feat + (size_t)(e0 + row) * FEAT + koff;
    v16h aE0 = load_a_f32(efr);
    v16h aE1 = load_a_f32(efr + 32);

    // h = relu(ef @ W1 + b1); re-layout C->A through LDS in two 16x32 halves
    v16h aH0 = {}, aH1 = {};
#pragma unroll
    for (int half = 0; half < 2; half++) {
#pragma unroll
      for (int nt2 = 0; nt2 < 2; nt2++) {
        const int nt = half * 2 + nt2;
        v8f z = {};
        v8f acc = wmma16(aE0, bload(sW1, nt, lane), z);
        acc = wmma16(aE1, bload(sW1, 4 + nt, lane), acc);
        const float bb = b1[nt * 16 + row];
#pragma unroll
        for (int r = 0; r < 8; r++) {
          float v = fmaxf(acc[r] + bb, 0.0f);
          myH[(r + moff) * 32 + nt2 * 16 + row] = (_Float16)v;  // row-major 16x32
        }
      }
      asm volatile("s_wait_dscnt 0x0" ::: "memory");
      v16h ah = load_a_lds(myH + row * 32 + koff);
      if (half == 0) aH0 = ah; else aH1 = ah;
    }

    // Gather source node features (node table is L2-resident: 12.8 MB)
    const int esrc = edge_idx[e0 + row];
    const float* srow = node_feat + (size_t)esrc * FEAT + koff;
    v16h aS0 = load_a_f32(srow);
    v16h aS1 = load_a_f32(srow + 32);

    int dstn[8];
#pragma unroll
    for (int r = 0; r < 8; r++) dstn[r] = edge_idx[NEDGES + e0 + r + moff];

#pragma unroll
    for (int k = 0; k < NK; k++) {
#pragma unroll
      for (int nt = 0; nt < 4; nt++) {
        v8f z = {};
        // edge weights for kernel k, cols [k*64+nt*16, +16)
        v8f aw = wmma16(aH0, bload(sW2, k * 4 + nt, lane), z);
        aw = wmma16(aH1, bload(sW2, 12 + k * 4 + nt, lane), aw);
        const float bb = b2[k * 64 + nt * 16 + row];
        // messages src @ W_k
        v8f am = wmma16(aS0, bload(sWk, k * 8 + nt, lane), z);
        am = wmma16(aS1, bload(sWk, k * 8 + 4 + nt, lane), am);
#pragma unroll
        for (int r = 0; r < 8; r++) {
          float m = am[r] * (aw[r] + bb);
          atomicAdd(sums + (size_t)dstn[r] * KU + k * 64 + nt * 16 + row, m);
        }
      }
    }
    if (lane < 16) atomicAdd(cnt + edge_idx[NEDGES + e0 + lane], 1.0f);
  }
}

// One wave per 16-node tile: mean -> fusion GEMM -> GroupNorm(16x4) -> relu.
__global__ __launch_bounds__(256) void node_out_kernel(
    const float* __restrict__ sums, const float* __restrict__ cnt,
    const float* __restrict__ FW, const float* __restrict__ fb,
    const float* __restrict__ gamma, const float* __restrict__ beta,
    float* __restrict__ out) {
  __shared__ __align__(64) _Float16 sFW[24 * 512];   // 24 KB: fusion_W 192x64
  const int tid = threadIdx.x;
  pack_b(sFW, FW, UNITS, 4, 24, tid, blockDim.x);
  __syncthreads();

  const int wave = tid >> 5, lane = tid & 31;
  const int row  = lane & 15;
  const int koff = (lane >> 4) << 3;
  const int moff = (lane >> 4) << 3;
  const int nTiles = NNODES / 16;
  const int tile = blockIdx.x * (blockDim.x >> 5) + wave;
  if (tile >= nTiles) return;  // wave-uniform

  const int n0 = tile << 4;
  const int node = n0 + row;
  const float invc = 1.0f / fmaxf(cnt[node], 1.0f);

  v16h aM[6];
#pragma unroll
  for (int kc = 0; kc < 6; kc++)
    aM[kc] = load_a_f32_scaled(sums + (size_t)node * KU + kc * 32 + koff, invc);

#pragma unroll
  for (int nt = 0; nt < 4; nt++) {
    v8f acc = {};
#pragma unroll
    for (int kc = 0; kc < 6; kc++)
      acc = wmma16(aM[kc], bload(sFW, kc * 4 + nt, lane), acc);

    const int col = nt * 16 + row;
    const float bb = fb[col], g = gamma[col], be = beta[col];
#pragma unroll
    for (int r = 0; r < 8; r++) {
      float v = acc[r] + bb;
      // GroupNorm: 4 consecutive cols per group = 4 adjacent lanes (same row)
      float s = v + __shfl_xor(v, 1, 32);
      s += __shfl_xor(s, 2, 32);
      float q = v * v;
      float sq = q + __shfl_xor(q, 1, 32);
      sq += __shfl_xor(sq, 2, 32);
      float mu  = s * 0.25f;
      float var = sq * 0.25f - mu * mu;
      float y = (v - mu) * rsqrtf(var + 1e-5f) * g + be;
      y = fmaxf(y, 0.0f);
      out[(size_t)(n0 + r + moff) * UNITS + col] = y;
    }
  }
}

// ---------------------------------------------------------------- launch

extern "C" void kernel_launch(void* const* d_in, const int* in_sizes, int n_in,
                              void* d_out, int out_size, void* d_ws, size_t ws_size,
                              hipStream_t stream) {
  const float* node_feat = (const float*)d_in[0];
  const int*   edge_idx  = (const int*)d_in[1];   // [2, E]: src then dst
  const float* edge_feat = (const float*)d_in[2];
  const float* Wk        = (const float*)d_in[3];
  const float* W1        = (const float*)d_in[4];
  const float* b1        = (const float*)d_in[5];
  const float* W2        = (const float*)d_in[6];
  const float* b2        = (const float*)d_in[7];
  const float* FW        = (const float*)d_in[8];
  const float* fb        = (const float*)d_in[9];
  const float* gamma     = (const float*)d_in[10];
  const float* beta      = (const float*)d_in[11];
  float* out = (float*)d_out;

  float* sums = (float*)d_ws;                 // [N, 192]
  float* cnt  = sums + (size_t)NNODES * KU;   // [N]

  zero_ws_kernel<<<2048, 256, 0, stream>>>(sums, NNODES * (KU + 1));
  edge_msg_kernel<<<1024, 128, 0, stream>>>(node_feat, edge_idx, edge_feat, Wk,
                                            W1, b1, W2, b2, sums, cnt);
  node_out_kernel<<<(NNODES / 16 + 7) / 8, 256, 0, stream>>>(sums, cnt, FW, fb,
                                                             gamma, beta, out);
}
